// ObsSwinEncoder_58222576664950
// MI455X (gfx1250) — matrix operand order repaired and checked by reading, not compiled, not run on hardware
//
#include <hip/hip_runtime.h>
#include <hip/hip_bf16.h>
#include <math.h>

typedef __attribute__((ext_vector_type(16))) __bf16 v16bf;
typedef __attribute__((ext_vector_type(8)))  float  v8f;
typedef __attribute__((ext_vector_type(8)))  unsigned short ushort8;

#define BATCH 256
#define IN_CH 32
#define HW_GRID 16              // h = w = 16 after 4x4 patching of 64x64
#define NTOK 256                // tokens per image
#define TTOK (BATCH * NTOK)     // 65536 total tokens
#define CDIM 96
#define QKVDIM 288
#define MLPH 384
#define PATCHK 512              // 32 * 4 * 4

// ---------------------------------------------------------------------------
// Patch matrix staging: x[B,32,64,64] -> bf16 A matrix [TTOK x 512]
// ---------------------------------------------------------------------------
__global__ void patch_stage_kernel(const float* __restrict__ x,
                                   __hip_bfloat16* __restrict__ out) {
    long idx = (long)blockIdx.x * blockDim.x + threadIdx.x;
    long total = (long)TTOK * PATCHK;
    if (idx >= total) return;
    int t  = (int)(idx >> 9);
    int k  = (int)(idx & 511);
    int b  = t >> 8;
    int r  = t & 255;
    int i  = r >> 4;
    int j  = r & 15;
    int ic = k >> 4;
    int ph = (k >> 2) & 3;
    int pw = k & 3;
    long src = (((long)b * IN_CH + ic) * 64 + (i * 4 + ph)) * 64 + (j * 4 + pw);
    out[idx] = __float2bfloat16(x[src]);
}

// ---------------------------------------------------------------------------
// Weight packing: f32 [K x N] (or [N x K] when srcT=1) -> bf16 fragment-major
// layout [N/16 tiles][K/32 ksteps][32 lanes][16 elems], so each lane's
// 16x16x32-bf16 B fragment is one contiguous 32-byte read.
//   lane 0-15  hold col = lane,    K = ks*32 + 0..15
//   lane 16-31 hold col = lane-16, K = ks*32 + 16..31
// ---------------------------------------------------------------------------
__global__ void w_pack_kernel(const float* __restrict__ src,
                              __hip_bfloat16* __restrict__ dst,
                              int N, int K, int srcT) {
    int idx = blockIdx.x * blockDim.x + threadIdx.x;
    if (idx >= N * K) return;
    int e    = idx & 15;
    int lane = (idx >> 4) & 31;
    int rem  = idx >> 9;
    int ksteps = K >> 5;
    int ks   = rem % ksteps;
    int tile = rem / ksteps;
    int col  = tile * 16 + (lane & 15);
    int kk   = ks * 32 + (lane >> 4) * 16 + e;
    float v = srcT ? src[(size_t)col * K + kk] : src[(size_t)kk * N + col];
    dst[idx] = __float2bfloat16(v);
}

// ---------------------------------------------------------------------------
// WMMA GEMM: Out[M x N] = A[M x K](bf16) * W(packed bf16) + bias
// One 16 x (16*TN) strip per wave. Software-pipelined: fragments for K-step
// ks+1 are loaded while the TN v_wmma_f32_16x16x32_bf16 of step ks execute,
// so waits become partial loadcnt waits instead of full drains.
// ACT: 0 = none, 1 = exact GELU.
// ---------------------------------------------------------------------------
template <int K, int TN, int ACT>
__global__ void gemm_wmma_kernel(const __hip_bfloat16* __restrict__ A,
                                 const __hip_bfloat16* __restrict__ Wp,
                                 const float* __restrict__ bias,
                                 float* __restrict__ outF,
                                 __hip_bfloat16* __restrict__ outB,
                                 int M, int N) {
    constexpr int KSTEPS = K / 32;
    int wave = blockIdx.x * (blockDim.x >> 5) + (threadIdx.x >> 5);
    int lane = threadIdx.x & 31;
    int stripsN = N / (16 * TN);
    int strips  = (M >> 4) * stripsN;
    if (wave >= strips) return;
    int tm = wave / stripsN;
    int sn = wave - tm * stripsN;
    int tile0 = sn * TN;

    int rowA = tm * 16 + (lane & 15);
    int kbA  = (lane >> 4) * 16;  // A: lanes 0-15 K 0-15, lanes 16-31 K 16-31
    const unsigned short* Ar = (const unsigned short*)A + (size_t)rowA * K + kbA;
    // B fragments: strip base + j*KSTEPS*512 + ks*512 + lane*16
    const unsigned short* Bb =
        (const unsigned short*)Wp + (size_t)tile0 * KSTEPS * 512 + lane * 16;

    v8f acc[TN];
#pragma unroll
    for (int j = 0; j < TN; ++j) acc[j] = (v8f){};

    v16bf afrag[2];
    v16bf bfrag[2][TN];

#define LOAD_A(ks, buf)                                                        \
    {                                                                          \
        ushort8 lo = *(const ushort8*)(Ar + (ks) * 32);                        \
        ushort8 hi = *(const ushort8*)(Ar + (ks) * 32 + 8);                    \
        unsigned short* ap = (unsigned short*)&afrag[buf];                     \
        _Pragma("unroll") for (int e = 0; e < 8; ++e) {                        \
            ap[e] = lo[e]; ap[8 + e] = hi[e];                                  \
        }                                                                      \
    }
#define LOAD_B(ks, j, buf)                                                     \
    {                                                                          \
        const unsigned short* bp = Bb + ((size_t)(j) * KSTEPS + (ks)) * 512;   \
        ushort8 lo = *(const ushort8*)(bp);                                    \
        ushort8 hi = *(const ushort8*)(bp + 8);                                \
        unsigned short* bq = (unsigned short*)&bfrag[buf][j];                  \
        _Pragma("unroll") for (int e = 0; e < 8; ++e) {                        \
            bq[e] = lo[e]; bq[8 + e] = hi[e];                                  \
        }                                                                      \
    }

    // prologue: fill buffer 0
    LOAD_A(0, 0);
#pragma unroll
    for (int j = 0; j < TN; ++j) LOAD_B(0, j, 0);

#pragma unroll
    for (int ks = 0; ks < KSTEPS; ++ks) {
        const int cur = ks & 1;
        const int nxt = cur ^ 1;
        if (ks + 1 < KSTEPS) {
            LOAD_A(ks + 1, nxt);
#pragma unroll
            for (int j = 0; j < TN; ++j) LOAD_B(ks + 1, j, nxt);
        }
#pragma unroll
        for (int j = 0; j < TN; ++j) {
            acc[j] = __builtin_amdgcn_wmma_f32_16x16x32_bf16(
                false, afrag[cur], false, bfrag[cur][j], (short)0, acc[j],
                false, false);
        }
    }
#undef LOAD_A
#undef LOAD_B

    int rbase = tm * 16 + (lane >> 4) * 8;   // C/D: VGPR v holds row v + 8*(lane>=16)
#pragma unroll
    for (int j = 0; j < TN; ++j) {
        int colB = (tile0 + j) * 16 + (lane & 15);
        float bcol = bias ? bias[colB] : 0.0f;
#pragma unroll
        for (int v = 0; v < 8; ++v) {
            float val = acc[j][v] + bcol;
            if (ACT == 1) val = 0.5f * val * (1.0f + erff(val * 0.70710678f));
            size_t o = (size_t)(rbase + v) * N + colB;
            if (outF) outF[o] = val;
            if (outB) outB[o] = __float2bfloat16(val);
        }
    }
}

// ---------------------------------------------------------------------------
// LayerNorm over C=96 per token; one wave per token (wave32, 3 elems/lane).
// res != null -> input is in+res. storePre: x_out gets pre-LN sum (residual
// update); else x_out gets the LN result. bf_out gets LN result in bf16.
// ---------------------------------------------------------------------------
__global__ void ln_kernel(const float* __restrict__ in,
                          const float* __restrict__ res,
                          float* __restrict__ x_out,
                          __hip_bfloat16* __restrict__ bf_out,
                          const float* __restrict__ g,
                          const float* __restrict__ bt,
                          int storePre) {
    int t = blockIdx.x * (blockDim.x >> 5) + (threadIdx.x >> 5);
    int lane = threadIdx.x & 31;
    if (t >= TTOK) return;
    const float* ip = in + (size_t)t * CDIM;
    float v0 = ip[lane], v1 = ip[lane + 32], v2 = ip[lane + 64];
    if (res) {
        const float* rp = res + (size_t)t * CDIM;
        v0 += rp[lane]; v1 += rp[lane + 32]; v2 += rp[lane + 64];
    }
    if (x_out && storePre) {
        float* xp = x_out + (size_t)t * CDIM;
        xp[lane] = v0; xp[lane + 32] = v1; xp[lane + 64] = v2;
    }
    float s = v0 + v1 + v2;
    float q = v0 * v0 + v1 * v1 + v2 * v2;
#pragma unroll
    for (int off = 16; off > 0; off >>= 1) {
        s += __shfl_xor(s, off, 32);
        q += __shfl_xor(q, off, 32);
    }
    float mu  = s * (1.0f / CDIM);
    float var = q * (1.0f / CDIM) - mu * mu;
    float inv = rsqrtf(var + 1e-5f);
    float o0 = (v0 - mu) * inv * g[lane]      + bt[lane];
    float o1 = (v1 - mu) * inv * g[lane + 32] + bt[lane + 32];
    float o2 = (v2 - mu) * inv * g[lane + 64] + bt[lane + 64];
    if (x_out && !storePre) {
        float* xp = x_out + (size_t)t * CDIM;
        xp[lane] = o0; xp[lane + 32] = o1; xp[lane + 64] = o2;
    }
    if (bf_out) {
        __hip_bfloat16* bp = bf_out + (size_t)t * CDIM;
        bp[lane]      = __float2bfloat16(o0);
        bp[lane + 32] = __float2bfloat16(o1);
        bp[lane + 64] = __float2bfloat16(o2);
    }
}

__global__ void add_kernel(float* __restrict__ x, const float* __restrict__ y,
                           long n) {
    long idx = (long)blockIdx.x * blockDim.x + threadIdx.x;
    if (idx < n) x[idx] += y[idx];
}

// ---------------------------------------------------------------------------
// Window attention: one wave per (window, head). 4096 windows x 4 heads.
// K/V staged in LDS; lanes 0-15 each own one query row; shift handled by
// address arithmetic and analytic mask (-100 across window-group boundaries).
// ---------------------------------------------------------------------------
__global__ void attn_kernel(const float* __restrict__ qkv,
                            __hip_bfloat16* __restrict__ out, int shift) {
    __shared__ float Ksh[8][16 * 24];
    __shared__ float Vsh[8][16 * 24];
    int wslot = threadIdx.x >> 5;
    int wave  = blockIdx.x * 8 + wslot;
    int lane  = threadIdx.x & 31;

    int head = wave & 3;
    int win  = wave >> 2;
    int b    = win >> 4;
    int wn   = win & 15;
    int wi   = wn >> 2, wj = wn & 3;

    for (int e = lane; e < 16 * 24; e += 32) {
        int tok = e / 24;
        int d   = e - tok * 24;
        int pi = tok >> 2, pj = tok & 3;
        int hh = (wi * 4 + pi + shift) & 15;
        int ww = (wj * 4 + pj + shift) & 15;
        size_t tb = ((size_t)b * NTOK + hh * HW_GRID + ww) * QKVDIM + head * 24 + d;
        Ksh[wslot][e] = qkv[tb + CDIM];
        Vsh[wslot][e] = qkv[tb + 2 * CDIM];
    }
    __syncthreads();

    if (lane < 16) {
        int q  = lane;
        int pi = q >> 2, pj = q & 3;
        int hh = (wi * 4 + pi + shift) & 15;
        int ww = (wj * 4 + pj + shift) & 15;
        size_t tq = (size_t)b * NTOK + hh * HW_GRID + ww;
        const float* qp = qkv + tq * QKVDIM + head * 24;
        float qv[24];
#pragma unroll
        for (int d = 0; d < 24; ++d) qv[d] = qp[d] * 0.20412414523f; // 24^-0.5
        int labq = (hh >> 2) * 4 + (ww >> 2);
        float sc[16];
        float mx = -1e30f;
        for (int k2 = 0; k2 < 16; ++k2) {
            float dot = 0.0f;
#pragma unroll
            for (int d = 0; d < 24; ++d) dot += qv[d] * Ksh[wslot][k2 * 24 + d];
            if (shift) {
                int ki = k2 >> 2, kj = k2 & 3;
                int kh = (wi * 4 + ki + shift) & 15;
                int kw = (wj * 4 + kj + shift) & 15;
                int labk = (kh >> 2) * 4 + (kw >> 2);
                if (labk != labq) dot -= 100.0f;
            }
            sc[k2] = dot;
            mx = fmaxf(mx, dot);
        }
        float se = 0.0f;
        for (int k2 = 0; k2 < 16; ++k2) { sc[k2] = __expf(sc[k2] - mx); se += sc[k2]; }
        float inv = 1.0f / se;
        __hip_bfloat16* op = out + tq * CDIM + head * 24;
        for (int d = 0; d < 24; ++d) {
            float o = 0.0f;
            for (int k2 = 0; k2 < 16; ++k2) o += sc[k2] * Vsh[wslot][k2 * 24 + d];
            op[d] = __float2bfloat16(o * inv);
        }
    }
}

// Mean pool over 256 tokens per image: bf16 [TTOK x 96] -> bf16 [256 x 96]
__global__ void pool_kernel(const __hip_bfloat16* __restrict__ in,
                            __hip_bfloat16* __restrict__ out) {
    int idx = blockIdx.x * blockDim.x + threadIdx.x;
    if (idx >= BATCH * CDIM) return;
    int b = idx / CDIM;
    int c = idx - b * CDIM;
    float s = 0.0f;
    const __hip_bfloat16* p = in + (size_t)b * NTOK * CDIM + c;
    for (int n = 0; n < NTOK; ++n) s += __bfloat162float(p[(size_t)n * CDIM]);
    out[idx] = __float2bfloat16(s * (1.0f / NTOK));
}

// ---------------------------------------------------------------------------
// Host-side orchestration
// ---------------------------------------------------------------------------
static inline int strip_blocks(int M, int N, int TN) {
    int strips = (M >> 4) * (N / (16 * TN));
    return (strips + 7) / 8;   // 8 waves (256 threads) per block
}

extern "C" void kernel_launch(void* const* d_in, const int* in_sizes, int n_in,
                              void* d_out, int out_size, void* d_ws, size_t ws_size,
                              hipStream_t stream) {
    (void)in_sizes; (void)n_in; (void)out_size; (void)ws_size;
    char* ws = (char*)d_ws;

    // workspace layout
    const size_t OFF_SA   = 0;                                   // bf16 [TTOK x 512]
    const size_t OFF_SB   = OFF_SA  + (size_t)TTOK * 512 * 2;    // bf16 [TTOK x 384]
    const size_t OFF_X    = OFF_SB  + (size_t)TTOK * MLPH * 2;   // f32  [TTOK x 96]
    const size_t OFF_Y    = OFF_X   + (size_t)TTOK * CDIM * 4;   // f32  [TTOK x 96]
    const size_t OFF_QKV  = OFF_Y   + (size_t)TTOK * CDIM * 4;   // f32  [TTOK x 288]
    const size_t OFF_W    = OFF_QKV + (size_t)TTOK * QKVDIM * 4; // bf16 weights (packed)
    const size_t OFF_POOL = OFF_W   + (size_t)294912 * 2;        // bf16 [256 x 96]

    __hip_bfloat16* SA   = (__hip_bfloat16*)(ws + OFF_SA);
    __hip_bfloat16* SB   = (__hip_bfloat16*)(ws + OFF_SB);
    float*          X    = (float*)(ws + OFF_X);
    float*          Y    = (float*)(ws + OFF_Y);
    float*          QKV  = (float*)(ws + OFF_QKV);
    __hip_bfloat16* WBF  = (__hip_bfloat16*)(ws + OFF_W);
    __hip_bfloat16* POOL = (__hip_bfloat16*)(ws + OFF_POOL);

    // packed bf16 weight region element offsets (element counts unchanged)
    const int W_CONV = 0;                         // 512*96  = 49152
    const int W_QKV0 = 49152;                     // 96*288  = 27648
    const int W_PRJ0 = 76800;                     // 96*96   = 9216
    const int W_FC10 = 86016;                     // 96*384  = 36864
    const int W_FC20 = 122880;                    // 384*96  = 36864
    const int W_QKV1 = 159744;
    const int W_PRJ1 = 187392;
    const int W_FC11 = 196608;
    const int W_FC21 = 233472;
    const int W_OUT  = 270336;                    // 96*256  = 24576

    const float* x_in    = (const float*)d_in[0];
    const float* conv_w  = (const float*)d_in[1];
    const float* conv_b  = (const float*)d_in[2];
    const float* pe_g    = (const float*)d_in[3];
    const float* pe_b    = (const float*)d_in[4];
    const float* norm_g  = (const float*)d_in[5];
    const float* norm_b  = (const float*)d_in[6];
    const float* out_w   = (const float*)d_in[7];
    const float* out_b   = (const float*)d_in[8];

    // --- prep: stage patches + pack weights into fragment-major bf16 ---------
    {
        long total = (long)TTOK * PATCHK;
        patch_stage_kernel<<<(int)((total + 255) / 256), 256, 0, stream>>>(x_in, SA);
    }
    // conv weight is [O=96 x 512] = [N x K] -> srcT = 1
    w_pack_kernel<<<(49152 + 255) / 256, 256, 0, stream>>>(conv_w, WBF + W_CONV, CDIM, PATCHK, 1);
    w_pack_kernel<<<(27648 + 255) / 256, 256, 0, stream>>>((const float*)d_in[11], WBF + W_QKV0, QKVDIM, CDIM, 0);
    w_pack_kernel<<<(9216  + 255) / 256, 256, 0, stream>>>((const float*)d_in[13], WBF + W_PRJ0, CDIM, CDIM, 0);
    w_pack_kernel<<<(36864 + 255) / 256, 256, 0, stream>>>((const float*)d_in[17], WBF + W_FC10, MLPH, CDIM, 0);
    w_pack_kernel<<<(36864 + 255) / 256, 256, 0, stream>>>((const float*)d_in[19], WBF + W_FC20, CDIM, MLPH, 0);
    w_pack_kernel<<<(27648 + 255) / 256, 256, 0, stream>>>((const float*)d_in[23], WBF + W_QKV1, QKVDIM, CDIM, 0);
    w_pack_kernel<<<(9216  + 255) / 256, 256, 0, stream>>>((const float*)d_in[25], WBF + W_PRJ1, CDIM, CDIM, 0);
    w_pack_kernel<<<(36864 + 255) / 256, 256, 0, stream>>>((const float*)d_in[29], WBF + W_FC11, MLPH, CDIM, 0);
    w_pack_kernel<<<(36864 + 255) / 256, 256, 0, stream>>>((const float*)d_in[31], WBF + W_FC21, CDIM, MLPH, 0);
    w_pack_kernel<<<(24576 + 255) / 256, 256, 0, stream>>>(out_w, WBF + W_OUT, 256, CDIM, 0);

    // --- patch embed GEMM + LN ----------------------------------------------
    gemm_wmma_kernel<PATCHK, 6, 0><<<strip_blocks(TTOK, CDIM, 6), 256, 0, stream>>>(
        SA, WBF + W_CONV, conv_b, Y, nullptr, TTOK, CDIM);
    ln_kernel<<<TTOK / 8, 256, 0, stream>>>(Y, nullptr, X, nullptr, pe_g, pe_b, 0);

    // --- swin blocks ---------------------------------------------------------
    for (int blk = 0; blk < 2; ++blk) {
        int base  = 9 + blk * 12;
        int shift = (blk & 1) ? 2 : 0;
        const float* n1_g   = (const float*)d_in[base + 0];
        const float* n1_b   = (const float*)d_in[base + 1];
        const float* qkv_b  = (const float*)d_in[base + 3];
        const float* proj_b = (const float*)d_in[base + 5];
        const float* n2_g   = (const float*)d_in[base + 6];
        const float* n2_b   = (const float*)d_in[base + 7];
        const float* fc1_b  = (const float*)d_in[base + 9];
        const float* fc2_b  = (const float*)d_in[base + 11];
        const __hip_bfloat16* wqkv = WBF + (blk ? W_QKV1 : W_QKV0);
        const __hip_bfloat16* wprj = WBF + (blk ? W_PRJ1 : W_PRJ0);
        const __hip_bfloat16* wfc1 = WBF + (blk ? W_FC11 : W_FC10);
        const __hip_bfloat16* wfc2 = WBF + (blk ? W_FC21 : W_FC20);

        // LN1 -> bf16 staging
        ln_kernel<<<TTOK / 8, 256, 0, stream>>>(X, nullptr, nullptr, SA, n1_g, n1_b, 0);
        // QKV GEMM (strips of 6 tiles: 288 = 3 strips)
        gemm_wmma_kernel<CDIM, 6, 0><<<strip_blocks(TTOK, QKVDIM, 6), 256, 0, stream>>>(
            SA, wqkv, qkv_b, QKV, nullptr, TTOK, QKVDIM);
        // window attention (4096 windows * 4 heads = 16384 waves)
        attn_kernel<<<2048, 256, 0, stream>>>(QKV, SA, shift);
        // proj GEMM
        gemm_wmma_kernel<CDIM, 6, 0><<<strip_blocks(TTOK, CDIM, 6), 256, 0, stream>>>(
            SA, wprj, proj_b, Y, nullptr, TTOK, CDIM);
        // residual add + LN2 (fused): X = X + Y ; SA = LN(X)
        ln_kernel<<<TTOK / 8, 256, 0, stream>>>(Y, X, X, SA, n2_g, n2_b, 1);
        // FC1 GEMM + GELU -> bf16 staging
        gemm_wmma_kernel<CDIM, 6, 1><<<strip_blocks(TTOK, MLPH, 6), 256, 0, stream>>>(
            SA, wfc1, fc1_b, nullptr, SB, TTOK, MLPH);
        // FC2 GEMM
        gemm_wmma_kernel<MLPH, 6, 0><<<strip_blocks(TTOK, CDIM, 6), 256, 0, stream>>>(
            SB, wfc2, fc2_b, Y, nullptr, TTOK, CDIM);
        // residual add
        add_kernel<<<(int)(((long)TTOK * CDIM + 255) / 256), 256, 0, stream>>>(
            X, Y, (long)TTOK * CDIM);
    }

    // --- final LN, mean pool, head GEMM -------------------------------------
    ln_kernel<<<TTOK / 8, 256, 0, stream>>>(X, nullptr, nullptr, SA, norm_g, norm_b, 0);
    pool_kernel<<<(BATCH * CDIM + 255) / 256, 256, 0, stream>>>(SA, POOL);
    gemm_wmma_kernel<CDIM, 4, 0><<<strip_blocks(BATCH, 256, 4), 256, 0, stream>>>(
        POOL, WBF + W_OUT, out_b, (float*)d_out, nullptr, BATCH, 256);
}
